// GNN_34256659153098
// MI455X (gfx1250) — compile-verified, matrix operand order checked by reading, not run on hardware
//
#include <hip/hip_runtime.h>
#include <hip/hip_bf16.h>
#include <stdint.h>

#define N_PTS  4096
#define C_DIM  512
#define OUT_DIM 256
#define BN_EPS 1e-5f

typedef __attribute__((ext_vector_type(16))) __bf16 v16bf;
typedef __attribute__((ext_vector_type(8)))  float  v8f;

union Frag {
    uint4 q[2];
    v16bf v;
};

__device__ __forceinline__ unsigned short f32_to_bf16(float f) {
    unsigned int u = __float_as_uint(f);
    u += 0x7FFFu + ((u >> 16) & 1u);          // round-to-nearest-even
    return (unsigned short)(u >> 16);
}

// ---------------------------------------------------------------------------
// Elementwise f32 -> bf16 convert
// ---------------------------------------------------------------------------
__global__ void cvt_bf16_kernel(const float* __restrict__ in,
                                unsigned short* __restrict__ out, int n) {
    int i = blockIdx.x * blockDim.x + threadIdx.x;
    if (i < n) out[i] = f32_to_bf16(in[i]);
}

// ---------------------------------------------------------------------------
// Tiled bf16 WMMA GEMM:  C[M,N] = A[M,K] @ B[N,K]^T (+ bias[N])
// Block: 256 threads = 8 waves, covers 128x128 of C.
// Wave (wm,wn): rows m0..m0+31 (2 M-tiles), cols n0..n0+63 (4 N-tiles).
// A-frag layout per ISA: lane<16 -> M=lane, K = kb+{0..7, 16..23};
//                        lane>=16 -> same M set, K = kb+{8..15, 24..31}.
// B mirrors A with N in place of M (B supplied row-major [N,K]).
// ---------------------------------------------------------------------------
__global__ __launch_bounds__(256)
void gemm_bf16_tn(const unsigned short* __restrict__ A,
                  const unsigned short* __restrict__ B,
                  const float* __restrict__ bias,
                  float* __restrict__ C,
                  int M, int N, int K) {
    const int lane = threadIdx.x & 31;
    const int wave = threadIdx.x >> 5;
    const int wm   = wave >> 1;                 // 0..3
    const int wn   = wave & 1;                  // 0..1
    const int m0   = blockIdx.y * 128 + wm * 32;
    const int n0   = blockIdx.x * 128 + wn * 64;
    const int mr   = lane & 15;
    const int sel  = lane >> 4;                 // 0 or 1

    v8f acc[2][4];
#pragma unroll
    for (int i = 0; i < 2; i++)
#pragma unroll
        for (int j = 0; j < 4; j++)
            acc[i][j] = (v8f){0.f,0.f,0.f,0.f,0.f,0.f,0.f,0.f};

    for (int kb = 0; kb < K; kb += 32) {
        Frag a[2], b[4];
#pragma unroll
        for (int i = 0; i < 2; i++) {
            const uint4* p = (const uint4*)(A + (size_t)(m0 + i*16 + mr) * K + kb);
            a[i].q[0] = p[sel];       // K = kb + sel*8 + 0..7
            a[i].q[1] = p[2 + sel];   // K = kb + 16 + sel*8 + 0..7
        }
#pragma unroll
        for (int j = 0; j < 4; j++) {
            const uint4* p = (const uint4*)(B + (size_t)(n0 + j*16 + mr) * K + kb);
            b[j].q[0] = p[sel];
            b[j].q[1] = p[2 + sel];
        }
#pragma unroll
        for (int i = 0; i < 2; i++)
#pragma unroll
            for (int j = 0; j < 4; j++)
                acc[i][j] = __builtin_amdgcn_wmma_f32_16x16x32_bf16(
                    false, a[i].v, false, b[j].v, (short)0, acc[i][j], false, false);
    }

    // C/D layout: VGPR r, lane<16 -> M = r; lane>=16 -> M = 8 + r; N = lane&15
#pragma unroll
    for (int i = 0; i < 2; i++) {
        const int rbase = m0 + i*16 + sel*8;
#pragma unroll
        for (int j = 0; j < 4; j++) {
            const int c  = n0 + j*16 + mr;
            const float bv = bias ? bias[c] : 0.0f;
#pragma unroll
            for (int r = 0; r < 8; r++)
                C[(size_t)(rbase + r) * N + c] = acc[i][j][r] + bv;
        }
    }
}

// ---------------------------------------------------------------------------
// Top-k threshold + degree:  one wave32 per row of si.
// thr[i] = 4th-largest of si[i,:]; dinv[i] = (#{j : si[i,j] >= thr[i]})^-1/2
// ---------------------------------------------------------------------------
__device__ __forceinline__ void top4_insert(float v, float& t0, float& t1,
                                            float& t2, float& t3) {
    if (v > t3) {
        if (v > t0)      { t3 = t2; t2 = t1; t1 = t0; t0 = v; }
        else if (v > t1) { t3 = t2; t2 = t1; t1 = v; }
        else if (v > t2) { t3 = t2; t2 = v; }
        else             { t3 = v; }
    }
}

__global__ __launch_bounds__(256)
void topk_kernel(const float* __restrict__ si,
                 float* __restrict__ thr, float* __restrict__ dinv) {
    const int lane = threadIdx.x & 31;
    const int row  = blockIdx.x * 8 + (threadIdx.x >> 5);
    const float* s = si + (size_t)row * N_PTS;

    float t0 = -3.4e38f, t1 = -3.4e38f, t2 = -3.4e38f, t3 = -3.4e38f;
    for (int j = lane; j < N_PTS; j += 32)
        top4_insert(s[j], t0, t1, t2, t3);

#pragma unroll
    for (int off = 16; off >= 1; off >>= 1) {
        float o0 = __shfl_down(t0, off, 32);
        float o1 = __shfl_down(t1, off, 32);
        float o2 = __shfl_down(t2, off, 32);
        float o3 = __shfl_down(t3, off, 32);
        top4_insert(o0, t0, t1, t2, t3);
        top4_insert(o1, t0, t1, t2, t3);
        top4_insert(o2, t0, t1, t2, t3);
        top4_insert(o3, t0, t1, t2, t3);
    }
    const float th = __shfl(t3, 0, 32);

    int cnt = 0;
    for (int j = lane; j < N_PTS; j += 32)
        cnt += (s[j] >= th) ? 1 : 0;
#pragma unroll
    for (int off = 16; off >= 1; off >>= 1)
        cnt += __shfl_down(cnt, off, 32);

    if (lane == 0) {
        thr[row]  = th;
        dinv[row] = rsqrtf((float)cnt);
    }
}

// ---------------------------------------------------------------------------
// Fused: sparse aggregate z = dinv_i * sum_{j in top-k row} dinv_j * v[j] + u[i],
// per-node BatchNorm (mean/var over C), residual + relu, store h as bf16.
// One 256-thread block per node; each thread owns channels t and t+256.
// ---------------------------------------------------------------------------
__global__ __launch_bounds__(256)
void agg_bn_kernel(const float* __restrict__ si, const float* __restrict__ thr,
                   const float* __restrict__ dinv, const float* __restrict__ v,
                   const float* __restrict__ u, const float* __restrict__ x,
                   unsigned short* __restrict__ hb) {
    const int row = blockIdx.x;
    const int t   = threadIdx.x;
    __shared__ int   nidx[128];
    __shared__ int   ncnt;
    __shared__ float rs[256], rq[256];

    if (t == 0) ncnt = 0;
    __syncthreads();

    const float  th   = thr[row];
    const float  di   = dinv[row];
    const float* srow = si + (size_t)row * N_PTS;
    for (int j = t; j < N_PTS; j += 256) {
        if (srow[j] >= th) {
            int p = atomicAdd(&ncnt, 1);
            if (p < 128) nidx[p] = j;
        }
    }
    __syncthreads();

    const int nn = ncnt < 128 ? ncnt : 128;   // typically exactly 4
    float s0 = 0.f, s1 = 0.f;
    for (int p = 0; p < nn; p++) {
        const int   j = nidx[p];
        const float w = dinv[j];
        s0 += w * v[(size_t)j * C_DIM + t];
        s1 += w * v[(size_t)j * C_DIM + t + 256];
    }
    const float z0 = di * s0 + u[(size_t)row * C_DIM + t];
    const float z1 = di * s1 + u[(size_t)row * C_DIM + t + 256];

    rs[t] = z0 + z1;
    rq[t] = z0 * z0 + z1 * z1;
    __syncthreads();
    for (int sft = 128; sft > 0; sft >>= 1) {
        if (t < sft) { rs[t] += rs[t + sft]; rq[t] += rq[t + sft]; }
        __syncthreads();
    }
    const float mean = rs[0] * (1.0f / C_DIM);
    const float var  = rq[0] * (1.0f / C_DIM) - mean * mean;
    const float inv  = rsqrtf(var + BN_EPS);

    float h0 = x[(size_t)row * C_DIM + t]       + (z0 - mean) * inv;
    float h1 = x[(size_t)row * C_DIM + t + 256] + (z1 - mean) * inv;
    h0 = h0 > 0.f ? h0 : 0.f;
    h1 = h1 > 0.f ? h1 : 0.f;
    hb[(size_t)row * C_DIM + t]       = f32_to_bf16(h0);
    hb[(size_t)row * C_DIM + t + 256] = f32_to_bf16(h1);
}

// ---------------------------------------------------------------------------
// Launch
// ---------------------------------------------------------------------------
extern "C" void kernel_launch(void* const* d_in, const int* in_sizes, int n_in,
                              void* d_out, int out_size, void* d_ws, size_t ws_size,
                              hipStream_t stream) {
    const float* x    = (const float*)d_in[0];   // [4096, 512]
    const float* U_w  = (const float*)d_in[1];   // [512, 512]
    const float* U_b  = (const float*)d_in[2];   // [512]
    const float* V_w  = (const float*)d_in[3];   // [512, 512]
    const float* V_b  = (const float*)d_in[4];   // [512]
    const float* fc_w = (const float*)d_in[5];   // [256, 512]
    const float* fc_b = (const float*)d_in[6];   // [256]
    float* out = (float*)d_out;                  // [4096, 256]

    char*  ws  = (char*)d_ws;
    size_t off = 0;
    auto alloc = [&](size_t bytes) -> char* {
        char* p = ws + off;
        off += (bytes + 255) & ~(size_t)255;
        return p;
    };
    unsigned short* xb   = (unsigned short*)alloc((size_t)N_PTS * C_DIM * 2);
    unsigned short* vwb  = (unsigned short*)alloc((size_t)C_DIM * C_DIM * 2);
    unsigned short* uwb  = (unsigned short*)alloc((size_t)C_DIM * C_DIM * 2);
    unsigned short* fwb  = (unsigned short*)alloc((size_t)OUT_DIM * C_DIM * 2);
    unsigned short* hb   = (unsigned short*)alloc((size_t)N_PTS * C_DIM * 2);
    float*          sib  = (float*)alloc((size_t)N_PTS * N_PTS * 4);
    float*          vbuf = (float*)alloc((size_t)N_PTS * C_DIM * 4);
    float*          ubuf = (float*)alloc((size_t)N_PTS * C_DIM * 4);
    float*          thr  = (float*)alloc((size_t)N_PTS * 4);
    float*          dnv  = (float*)alloc((size_t)N_PTS * 4);

    // 1) bf16 converts
    {
        int n;
        n = N_PTS * C_DIM;
        cvt_bf16_kernel<<<(n + 255) / 256, 256, 0, stream>>>(x, xb, n);
        n = C_DIM * C_DIM;
        cvt_bf16_kernel<<<(n + 255) / 256, 256, 0, stream>>>(V_w, vwb, n);
        cvt_bf16_kernel<<<(n + 255) / 256, 256, 0, stream>>>(U_w, uwb, n);
        n = OUT_DIM * C_DIM;
        cvt_bf16_kernel<<<(n + 255) / 256, 256, 0, stream>>>(fc_w, fwb, n);
    }

    // 2) v = x @ V_w^T + V_b ; u = x @ U_w^T + U_b
    {
        dim3 grid(C_DIM / 128, N_PTS / 128);
        gemm_bf16_tn<<<grid, 256, 0, stream>>>(xb, vwb, V_b, vbuf, N_PTS, C_DIM, C_DIM);
        gemm_bf16_tn<<<grid, 256, 0, stream>>>(xb, uwb, U_b, ubuf, N_PTS, C_DIM, C_DIM);
    }

    // 3) si = x @ x^T
    {
        dim3 grid(N_PTS / 128, N_PTS / 128);
        gemm_bf16_tn<<<grid, 256, 0, stream>>>(xb, xb, nullptr, sib, N_PTS, N_PTS, C_DIM);
    }

    // 4) top-k threshold + degree normalization
    topk_kernel<<<N_PTS / 8, 256, 0, stream>>>(sib, thr, dnv);

    // 5) sparse aggregate + BN + residual + relu -> h (bf16)
    agg_bn_kernel<<<N_PTS, 256, 0, stream>>>(sib, thr, dnv, vbuf, ubuf, x, hb);

    // 6) out = h @ fc_w^T + fc_b
    {
        dim3 grid(OUT_DIM / 128, N_PTS / 128);
        gemm_bf16_tn<<<grid, 256, 0, stream>>>(hb, fwb, fc_b, out, N_PTS, OUT_DIM, C_DIM);
    }
}
